// MinGRUCell_20916490731572
// MI455X (gfx1250) — matrix operand order repaired and checked by reading, not compiled, not run on hardware
//
#include <hip/hip_runtime.h>
#include <math.h>

#define BATCH 8
#define SEQ   4096
#define DIN_  1024
#define DH_   1024
#define MTOT  (BATCH * SEQ)      // 32768 rows of the GEMM
#define NCHUNK 16
#define TCHUNK (SEQ / NCHUNK)    // 256 steps per scan chunk

#define LDA 40                   // padded LDS row stride (bf16 elems) for A
#define LDB 40                   // padded LDS row stride (bf16 elems) for B

typedef __attribute__((ext_vector_type(16))) __bf16 bf16x16;
typedef __attribute__((ext_vector_type(8)))  float  f32x8;

// ---------------------------------------------------------------------------
// helpers
// ---------------------------------------------------------------------------
__device__ __forceinline__ float softplusf(float x) {
    return (x > 0.f) ? (x + log1pf(expf(-x))) : log1pf(expf(x));
}

union Bf4 { __bf16 h[4]; uint2 u; };
union FragU { bf16x16 v; uint4 q[2]; };

__device__ __forceinline__ uint2 cvt4(float4 f) {
    Bf4 r;
    r.h[0] = (__bf16)f.x; r.h[1] = (__bf16)f.y;
    r.h[2] = (__bf16)f.z; r.h[3] = (__bf16)f.w;
    return r.u;
}

// ---------------------------------------------------------------------------
// Kernel 1: fused GEMM (LDS-staged bf16 WMMA) + minGRU log-space epilogue
//   gh = X @ W^T + b ; gate = gh[:, :DH], hidden = gh[:, DH:]
//   logc = -softplus(gate) ; logv = -softplus(-gate) + log_g(hidden)
// WG = 8 waves (4 M-waves x 2 ch-waves) -> tile 256M x 64ch (128 W rows).
// Per K-step(32): cooperative global->bf16->LDS staging (double buffered),
// fragments fed via ds_load_b128, 16 WMMAs per wave.
// ---------------------------------------------------------------------------
__global__ __launch_bounds__(256)
void mingru_gemm_wmma(const float* __restrict__ X,   // (MTOT, DIN)
                      const float* __restrict__ W,   // (2*DH, DIN)
                      const float* __restrict__ bias,// (2*DH)
                      float* __restrict__ logc,      // (MTOT, DH)
                      float* __restrict__ logv)      // (MTOT, DH)
{
    __shared__ __bf16 As[2][256 * LDA];   // 2 x 20 KB
    __shared__ __bf16 Bs[2][128 * LDB];   // 2 x 10 KB

    const int tid   = threadIdx.x;
    const int lane  = tid & 31;
    const int w     = tid >> 5;
    const int wm    = w & 3;            // 0..3 : M wave
    const int wc    = w >> 2;           // 0..1 : ch wave
    const int m0blk = blockIdx.x * 256;
    const int cblk  = blockIdx.y;       // 64-channel block

    const int lrow  = lane & 15;        // A: row in tile / B,C: column in tile
    const int ahalf = (lane >> 4) * 8;  // A K-chunk base (0 or 8 elems)
    const int bhalf = (lane >> 4) * 8;  // B K-chunk base (0 or 8 elems)

    // accumulators: [M tile][ch tile] for gate / hidden halves
    f32x8 accg[4][2], acch[4][2];
#pragma unroll
    for (int mi = 0; mi < 4; ++mi)
#pragma unroll
        for (int ci = 0; ci < 2; ++ci) { accg[mi][ci] = (f32x8){}; acch[mi][ci] = (f32x8){}; }

    // staging: A = 256 rows x 8 float4-chunks, B = 128 rows x 8 chunks
    // (rows 0-63 gate, 64-127 hidden); 8 threads per row -> coalesced.
    const int acc_ = tid & 7;           // chunk index within row

    // prologue: stage k-slab 0 into buffer 0
    {
#pragma unroll
        for (int i = 0; i < 8; ++i) {
            const int row = (tid + i * 256) >> 3;
            float4 v = *(const float4*)(X + (size_t)(m0blk + row) * DIN_ + acc_ * 4);
            *(uint2*)&As[0][row * LDA + acc_ * 4] = cvt4(v);
        }
#pragma unroll
        for (int i = 0; i < 4; ++i) {
            const int row = (tid + i * 256) >> 3;
            const int wr  = (row < 64) ? (cblk * 64 + row) : (DH_ + cblk * 64 + (row - 64));
            float4 v = *(const float4*)(W + (size_t)wr * DIN_ + acc_ * 4);
            *(uint2*)&Bs[0][row * LDB + acc_ * 4] = cvt4(v);
        }
    }
    __syncthreads();

    const int NSTEP = DIN_ / 32;   // 32
    for (int s = 0; s < NSTEP; ++s) {
        const int cur = s & 1;
        const int nxt = cur ^ 1;
        const bool has_next = (s + 1 < NSTEP);
        const int knext = (s + 1) * 32;

        // ---- issue next-slab global loads; convert immediately so only
        //      packed bf16 (uint2 each) stays live across the WMMA block.
        uint2 sa[8], sb[4];
        if (has_next) {
            float4 ra[8], rb[4];
#pragma unroll
            for (int i = 0; i < 8; ++i) {
                const int row = (tid + i * 256) >> 3;
                ra[i] = *(const float4*)(X + (size_t)(m0blk + row) * DIN_ + knext + acc_ * 4);
            }
#pragma unroll
            for (int i = 0; i < 4; ++i) {
                const int row = (tid + i * 256) >> 3;
                const int wr  = (row < 64) ? (cblk * 64 + row) : (DH_ + cblk * 64 + (row - 64));
                rb[i] = *(const float4*)(W + (size_t)wr * DIN_ + knext + acc_ * 4);
            }
#pragma unroll
            for (int i = 0; i < 8; ++i) sa[i] = cvt4(ra[i]);
#pragma unroll
            for (int i = 0; i < 4; ++i) sb[i] = cvt4(rb[i]);
        }

        // ---- B fragments + first half of A; 8 WMMAs
        FragU fbg[2], fbh[2];
#pragma unroll
        for (int ci = 0; ci < 2; ++ci) {
            const int rg = wc * 32 + 16 * ci + lrow;       // gate row in Bs
            fbg[ci].q[0] = *(const uint4*)&Bs[cur][rg * LDB + bhalf];
            fbg[ci].q[1] = *(const uint4*)&Bs[cur][rg * LDB + bhalf + 16];
            const int rh = 64 + rg;                        // hidden row in Bs
            fbh[ci].q[0] = *(const uint4*)&Bs[cur][rh * LDB + bhalf];
            fbh[ci].q[1] = *(const uint4*)&Bs[cur][rh * LDB + bhalf + 16];
        }
        {
            FragU fa[2];
#pragma unroll
            for (int mi = 0; mi < 2; ++mi) {
                const int row = wm * 64 + 16 * mi + lrow;
                fa[mi].q[0] = *(const uint4*)&As[cur][row * LDA + ahalf];
                fa[mi].q[1] = *(const uint4*)&As[cur][row * LDA + 16 + ahalf];
            }
#pragma unroll
            for (int mi = 0; mi < 2; ++mi)
#pragma unroll
                for (int ci = 0; ci < 2; ++ci) {
                    accg[mi][ci] = __builtin_amdgcn_wmma_f32_16x16x32_bf16(
                        false, fa[mi].v, false, fbg[ci].v, (short)0, accg[mi][ci], false, false);
                    acch[mi][ci] = __builtin_amdgcn_wmma_f32_16x16x32_bf16(
                        false, fa[mi].v, false, fbh[ci].v, (short)0, acch[mi][ci], false, false);
                }
        }
        // ---- second half of A; 8 WMMAs
        {
            FragU fa[2];
#pragma unroll
            for (int mi = 0; mi < 2; ++mi) {
                const int row = wm * 64 + 16 * (mi + 2) + lrow;
                fa[mi].q[0] = *(const uint4*)&As[cur][row * LDA + ahalf];
                fa[mi].q[1] = *(const uint4*)&As[cur][row * LDA + 16 + ahalf];
            }
#pragma unroll
            for (int mi = 0; mi < 2; ++mi)
#pragma unroll
                for (int ci = 0; ci < 2; ++ci) {
                    accg[mi + 2][ci] = __builtin_amdgcn_wmma_f32_16x16x32_bf16(
                        false, fa[mi].v, false, fbg[ci].v, (short)0, accg[mi + 2][ci], false, false);
                    acch[mi + 2][ci] = __builtin_amdgcn_wmma_f32_16x16x32_bf16(
                        false, fa[mi].v, false, fbh[ci].v, (short)0, acch[mi + 2][ci], false, false);
                }
        }

        // ---- store next slab into the other buffer (after WMMAs so the
        //      matrix pipe hides the global-load latency)
        if (has_next) {
#pragma unroll
            for (int i = 0; i < 8; ++i) {
                const int row = (tid + i * 256) >> 3;
                *(uint2*)&As[nxt][row * LDA + acc_ * 4] = sa[i];
            }
#pragma unroll
            for (int i = 0; i < 4; ++i) {
                const int row = (tid + i * 256) >> 3;
                *(uint2*)&Bs[nxt][row * LDB + acc_ * 4] = sb[i];
            }
        }
        __syncthreads();
    }

    // ---- epilogue: bias + log-space transforms, direct store
    const int m0 = m0blk + wm * 64;
    const int c0 = cblk * 64 + wc * 32;
#pragma unroll
    for (int ci = 0; ci < 2; ++ci) {
        const int ch  = c0 + 16 * ci + lrow;     // C layout: N = lane & 15
        const float bgs = bias[ch];
        const float bhs = bias[DH_ + ch];
#pragma unroll
        for (int mi = 0; mi < 4; ++mi) {
#pragma unroll
            for (int r = 0; r < 8; ++r) {
                const int m = m0 + 16 * mi + r + ((lane >> 4) << 3);
                const float g  = accg[mi][ci][r] + bgs;
                const float hd = acch[mi][ci][r] + bhs;
                const float lc = -softplusf(g);            // log(1 - z)
                const float lz = -softplusf(-g);           // log z
                const float lg = (hd >= 0.f) ? logf(hd + 0.5f) : -softplusf(-hd);
                const size_t o = (size_t)m * DH_ + ch;
                logc[o] = lc;
                logv[o] = lz + lg;
            }
        }
    }
}

// ---------------------------------------------------------------------------
// Kernel 2: per-chunk reduction for the Heinsen log-space scan.
// ---------------------------------------------------------------------------
__global__ __launch_bounds__(256)
void mingru_scan_reduce(const float* __restrict__ logc,
                        const float* __restrict__ logv,
                        float* __restrict__ cA, float* __restrict__ cL)
{
    const int gid = blockIdx.x * blockDim.x + threadIdx.x;   // < BATCH*NCHUNK*DH
    const int ch  = gid % DH_;
    const int t0  = gid / DH_;
    const int c   = t0 % NCHUNK;
    const int b   = t0 / NCHUNK;

    size_t base = ((size_t)b * SEQ + (size_t)c * TCHUNK) * DH_ + ch;
    float a = 0.f, mm = -INFINITY, ss = 0.f;
    for (int t = 0; t < TCHUNK; ++t) {
        const float lc = logc[base];
        const float lv = logv[base];
        base += DH_;
        a += lc;
        const float u = lv - a;
        if (u > mm) { ss = ss * expf(mm - u) + 1.f; mm = u; }
        else        { ss += expf(u - mm); }
    }
    const int so = (b * DH_ + ch) * NCHUNK + c;
    cA[so] = a;
    cL[so] = mm + logf(ss);
}

// ---------------------------------------------------------------------------
// Kernel 3: combine chunk summaries per channel -> per-chunk start state.
// ---------------------------------------------------------------------------
__global__ __launch_bounds__(256)
void mingru_scan_combine(const float* __restrict__ h0,
                         const float* __restrict__ cA, const float* __restrict__ cL,
                         float* __restrict__ sA, float* __restrict__ sG)
{
    const int gid = blockIdx.x * blockDim.x + threadIdx.x;   // < BATCH*DH
    const int ch  = gid % DH_;
    const int b   = gid / DH_;

    float G  = logf(h0[(size_t)b * DH_ + ch]);   // h0 strictly positive
    float Ap = 0.f;
    const int sbase = (b * DH_ + ch) * NCHUNK;
    for (int c = 0; c < NCHUNK; ++c) {
        sA[sbase + c] = Ap;
        sG[sbase + c] = G;
        const float term = cL[sbase + c] - Ap;
        const float mx   = fmaxf(G, term);
        G  = mx + logf(expf(G - mx) + expf(term - mx));
        Ap += cA[sbase + c];
    }
}

// ---------------------------------------------------------------------------
// Kernel 4: apply pass — re-walk each chunk with its start state, emit h_t.
// ---------------------------------------------------------------------------
__global__ __launch_bounds__(256)
void mingru_scan_apply(const float* __restrict__ logc,
                       const float* __restrict__ logv,
                       const float* __restrict__ sA, const float* __restrict__ sG,
                       float* __restrict__ out)
{
    const int gid = blockIdx.x * blockDim.x + threadIdx.x;   // < BATCH*NCHUNK*DH
    const int ch  = gid % DH_;
    const int t0  = gid / DH_;
    const int c   = t0 % NCHUNK;
    const int b   = t0 / NCHUNK;

    const int so = (b * DH_ + ch) * NCHUNK + c;
    float a  = sA[so];
    float mm = sG[so];     // running logsumexp carried in (mm, ss)
    float ss = 1.f;

    size_t base = ((size_t)b * SEQ + (size_t)c * TCHUNK) * DH_ + ch;
    for (int t = 0; t < TCHUNK; ++t) {
        const float lc = logc[base];
        const float lv = logv[base];
        a += lc;
        const float u = lv - a;
        if (u > mm) { ss = ss * expf(mm - u) + 1.f; mm = u; }
        else        { ss += expf(u - mm); }
        out[base] = expf(a + mm + logf(ss));
        base += DH_;
    }
}

// ---------------------------------------------------------------------------
// launch
// ---------------------------------------------------------------------------
extern "C" void kernel_launch(void* const* d_in, const int* in_sizes, int n_in,
                              void* d_out, int out_size, void* d_ws, size_t ws_size,
                              hipStream_t stream)
{
    const float* x  = (const float*)d_in[0];   // (B, S, DIN)
    const float* h  = (const float*)d_in[1];   // (B, 1, DH)
    const float* W  = (const float*)d_in[2];   // (2*DH, DIN)
    const float* b  = (const float*)d_in[3];   // (2*DH)
    float* out      = (float*)d_out;           // (B, S, DH)

    char* ws = (char*)d_ws;
    const size_t n_md  = (size_t)MTOT * DH_;          // 33,554,432
    const size_t n_sum = (size_t)BATCH * DH_ * NCHUNK;

    float* logc = (float*)(ws);
    float* logv = (float*)(ws + n_md * 4);
    float* cA   = (float*)(ws + 2 * n_md * 4);
    float* cL   = (float*)(ws + 2 * n_md * 4 + n_sum * 4);
    float* sA   = (float*)(ws + 2 * n_md * 4 + 2 * n_sum * 4);
    float* sG   = (float*)(ws + 2 * n_md * 4 + 3 * n_sum * 4);

    // 1) fused GEMM + log epilogue
    dim3 ggrid(MTOT / 256, DH_ / 64);
    mingru_gemm_wmma<<<ggrid, 256, 0, stream>>>(x, W, b, logc, logv);

    // 2) chunk reduce: BATCH*NCHUNK*DH = 131072 threads
    mingru_scan_reduce<<<(BATCH * NCHUNK * DH_) / 256, 256, 0, stream>>>(logc, logv, cA, cL);

    // 3) combine: BATCH*DH = 8192 threads
    mingru_scan_combine<<<(BATCH * DH_) / 256, 256, 0, stream>>>(h, cA, cL, sA, sG);

    // 4) apply: 131072 threads
    mingru_scan_apply<<<(BATCH * NCHUNK * DH_) / 256, 256, 0, stream>>>(logc, logv, sA, sG, out);
}
